// Fused_height_14035953123677
// MI455X (gfx1250) — matrix operand order repaired and checked by reading, not compiled, not run on hardware
//
#include <hip/hip_runtime.h>
#include <stdint.h>

// Problem constants
#define Bm 8
#define Cch 128
#define Hh 128
#define Ww 128
#define HW (Hh * Ww)        // 16384
#define K1 1152             // 9*128 : deformable-conv GEMM depth
#define K1B 1280            // + 128 for fused w_bb 1x1 tail
#define K2 256              // concat 1x1 GEMM depth

// Workspace layout (bytes)
#define WS_WREGBB 0                       // 128*1280 bf16 = 327680
#define WS_WSHARE 327680                  // 128*256  bf16 =  65536
#define WS_WOFFM  393216                  // 27*1152  bf16 =  62208
#define WS_OFFS   455424                  // 8*18*16384 f32 = 9437184
#define WS_MASK   9892608                 // 8*9*16384  f32 = 4718592

typedef __attribute__((ext_vector_type(16))) __bf16 v16bf;
typedef __attribute__((ext_vector_type(8)))  float  v8f;
typedef unsigned int u32x4 __attribute__((ext_vector_type(4)));
typedef float        f32x4 __attribute__((ext_vector_type(4)));

union Frag { v16bf v; u32x4 q[2]; };

__device__ __forceinline__ unsigned short f2bf(float f) {
    unsigned u = __float_as_uint(f);
    u += 0x7FFFu + ((u >> 16) & 1u);      // round-to-nearest-even
    return (unsigned short)(u >> 16);
}
__device__ __forceinline__ float bf2f(unsigned short h) {
    return __uint_as_float(((unsigned)h) << 16);
}
__device__ __forceinline__ float sigmoidf_(float x) { return 1.0f / (1.0f + __expf(-x)); }

// ---------------------------------------------------------------------------
// Kernel 0: one-time fp32 -> bf16 weight packing into WMMA-friendly layouts.
//   wregbb[cout][k]  k<1152: w_reg (cin*9+point order, its native order),
//                    k>=1152: w_bb  (fused 1x1 tail)
//   wshare[cout][k]  k<128: s-half, k>=128: shared-half (matches concat order)
//   woffm [o][k]     o<18: w_off, o>=18: w_mask
// ---------------------------------------------------------------------------
__global__ void pack_weights_kernel(const float* __restrict__ w_off,
                                    const float* __restrict__ w_mask,
                                    const float* __restrict__ w_reg,
                                    const float* __restrict__ w_bb,
                                    const float* __restrict__ w_share,
                                    unsigned short* __restrict__ wregbb,
                                    unsigned short* __restrict__ wshare,
                                    unsigned short* __restrict__ woffm) {
    const int N0 = 128 * 1280, N1 = 128 * 256, N2 = 27 * 1152;
    int i = blockIdx.x * 256 + threadIdx.x;
    if (i < N0) {
        int cout = i / 1280, k = i - cout * 1280;
        float f = (k < K1) ? w_reg[cout * K1 + k] : w_bb[cout * 128 + (k - K1)];
        wregbb[i] = f2bf(f);
    } else if (i < N0 + N1) {
        int j = i - N0;
        wshare[j] = f2bf(w_share[j]);
    } else if (i < N0 + N1 + N2) {
        int j = i - N0 - N1;
        woffm[j] = f2bf((j < 18 * 1152) ? w_off[j] : w_mask[j - 18 * 1152]);
    }
}

// ---------------------------------------------------------------------------
// Kernel 1: offsets/mask 3x3 conv over DSMg (recomputed on the fly).
// N=27 is far below a WMMA tile and only ~11% of total FLOPs -> VALU FMA with
// all 27x1152 bf16 weights staged in LDS (62 KB, broadcast to all lanes).
// One thread per output pixel.
// ---------------------------------------------------------------------------
__global__ __launch_bounds__(256)
void offsets_mask_kernel(const float* __restrict__ HSI,
                         const float* __restrict__ DSM,
                         const float* __restrict__ b_off,
                         const float* __restrict__ b_mask,
                         const float* __restrict__ lam1,
                         const float* __restrict__ lam2,
                         const unsigned short* __restrict__ woffm,
                         float* __restrict__ offs,
                         float* __restrict__ mask) {
    __shared__ unsigned short wlds[27 * 1152];
    const int tid = threadIdx.x;
    for (int i = tid; i < 27 * 1152; i += 256) wlds[i] = woffm[i];
    __syncthreads();

    const int p   = blockIdx.x * 256 + tid;   // pixel id, exact coverage
    const int b   = p >> 14;                  // / 16384
    const int rem = p & 16383;
    const int y   = rem >> 7, x = rem & 127;
    const float s1 = sigmoidf_(lam1[0]);
    const float s2 = sigmoidf_(lam2[0]);

    float acc[27];
#pragma unroll
    for (int o = 0; o < 18; ++o) acc[o] = b_off[o];
#pragma unroll
    for (int o = 0; o < 9; ++o) acc[18 + o] = b_mask[o];

    for (int cin = 0; cin < 128; ++cin) {
        const size_t base = (size_t)(b * 128 + cin) * HW + (size_t)y * Ww + x;
        const float* dp = DSM + base;
        const float* hp = HSI + base;
        float g[9];
#pragma unroll
        for (int t = 0; t < 9; ++t) {              // zero-padded 3x3 window of DSMg
            int dy = t / 3 - 1, dx = t % 3 - 1;
            int yy = y + dy, xx = x + dx;
            bool in = ((unsigned)yy < (unsigned)Hh) && ((unsigned)xx < (unsigned)Ww);
            int off = dy * Ww + dx;
            g[t] = in ? (dp[off] * s1 + hp[off] * s2) : 0.0f;
        }
        const unsigned short* wr = wlds + cin * 9;
#pragma unroll
        for (int o = 0; o < 27; ++o) {
#pragma unroll
            for (int t = 0; t < 9; ++t)
                acc[o] = fmaf(bf2f(wr[o * 1152 + t]), g[t], acc[o]);
        }
    }
#pragma unroll
    for (int o = 0; o < 18; ++o)
        offs[(size_t)(b * 18 + o) * HW + (size_t)y * Ww + x] = acc[o];
#pragma unroll
    for (int kk = 0; kk < 9; ++kk)
        mask[(size_t)(b * 9 + kk) * HW + (size_t)y * Ww + x] = sigmoidf_(acc[18 + kk]);
}

// ---------------------------------------------------------------------------
// Kernel 2: fully fused deform-conv + ReLU + w_bb 1x1 + concat-1x1 + bias+ReLU.
// One workgroup = 16 pixels (x-segment of a row) x all 128 output channels.
// 8 waves; wave w owns output-channel tile [16w, 16w+16).
//   GEMM1: D[pix(16) x cout(16)] = A1[pix x 1152] * Wreg, bf16 WMMA, fp32 acc
//          -> ReLU -> continue accumulating A1[pix,1152:1280]*Wbb (+b_bb)
//   GEMM2: out = ReLU(A2[pix x 256] * Wshare + b_share), A2 = [s | shared]
// A-fragment: 16-bit A layout (lane&15 = M; K halves 0-7/8-15 by lane>>4, +16
// in upper 4 VGPRs). B-fragment: same K split as A (K=8h+s, +16 for s>=8),
// N=lane&15 -> two 16B loads (k0+8h, k0+16+8h) from k-major packed weights.
// ---------------------------------------------------------------------------
__global__ __launch_bounds__(256)
void deform_fused_kernel(const float* __restrict__ HSI,
                         const float* __restrict__ DSM,
                         const float* __restrict__ shared_in,
                         const float* __restrict__ b_bb,
                         const float* __restrict__ b_share,
                         const float* __restrict__ lam1,
                         const float* __restrict__ lam2,
                         const unsigned short* __restrict__ wregbb,
                         const unsigned short* __restrict__ wshare,
                         const float* __restrict__ offs,
                         const float* __restrict__ mask,
                         float* __restrict__ out) {
    __shared__ __align__(16) unsigned short A1[16 * K1B];  // 40960 B, bf16 GEMM1 A
    __shared__ __align__(16) unsigned short A2[16 * K2];   //  8192 B, bf16 GEMM2 A
    __shared__ float cw[144][4];                           // bilinear weights*mask
    __shared__ int   coff[144][4];                         // clipped flat offsets

    const int tid  = threadIdx.x;
    const int lane = tid & 31;
    const int wave = tid >> 5;
    const int half = lane >> 4;
    const int lo   = lane & 15;

    const int wg  = blockIdx.x;
    const int seg = wg & 7;                 // W/16 segments
    const int y   = (wg >> 3) & 127;
    const int b   = wg >> 10;
    const int x0  = seg * 16;

    const float s1 = sigmoidf_(lam1[0]);
    const float s2 = sigmoidf_(lam2[0]);

    // --- stage 1a: bilinear sampling coords for 9 points x 16 pixels ---------
    if (tid < 144) {
        const int point = tid >> 4, pix = tid & 15;
        const int xi = x0 + pix;
        const size_t sp = (size_t)y * Ww + xi;
        const float dy = offs[(size_t)(b * 18 + 2 * point)     * HW + sp];
        const float dx = offs[(size_t)(b * 18 + 2 * point + 1) * HW + sp];
        const float m  = mask[(size_t)(b * 9 + point) * HW + sp];
        const int i = point / 3, j = point % 3;
        const float fy = (float)y  + (float)(i - 1) + dy;
        const float fx = (float)xi + (float)(j - 1) + dx;
        const float y0f = floorf(fy), x0f = floorf(fx);
        const float wy = fy - y0f, wx = fx - x0f;
        const float cwt[4] = {(1.f - wy) * (1.f - wx), (1.f - wy) * wx,
                              wy * (1.f - wx),         wy * wx};
#pragma unroll
        for (int c = 0; c < 4; ++c) {
            const float yc = y0f + (float)(c >> 1);
            const float xc = x0f + (float)(c & 1);
            const bool valid = (yc >= 0.f) && (yc <= (float)(Hh - 1)) &&
                               (xc >= 0.f) && (xc <= (float)(Ww - 1));
            const int iy = (int)fminf(fmaxf(yc, 0.f), (float)(Hh - 1));
            const int ix = (int)fminf(fmaxf(xc, 0.f), (float)(Ww - 1));
            cw[tid][c]   = valid ? cwt[c] * m : 0.0f;
            coff[tid][c] = iy * Ww + ix;
        }
    }

    // --- stage 1b: DSMg tail of A1 (k=1152..1279) and shared tail of A2 ------
    for (int idx = tid; idx < 128 * 16; idx += 256) {
        const int ch = idx >> 4, pix = idx & 15;
        const size_t g = (size_t)(b * 128 + ch) * HW + (size_t)y * Ww + x0 + pix;
        A1[pix * K1B + K1 + ch] = f2bf(DSM[g] * s1 + HSI[g] * s2);
        A2[pix * K2 + 128 + ch] = f2bf(shared_in[g]);
    }
    __syncthreads();

    // --- stage 2: modulated bilinear gather -> A1[pix][cin*9+point] ----------
    for (int idx = tid; idx < 16 * K1; idx += 256) {
        const int pix = idx / K1;
        const int k   = idx - pix * K1;
        const int cin = k / 9;
        const int point = k - cin * 9;
        const float* hp = HSI + (size_t)(b * 128 + cin) * HW;
        const int q = point * 16 + pix;
        const float v = cw[q][0] * hp[coff[q][0]] + cw[q][1] * hp[coff[q][1]] +
                        cw[q][2] * hp[coff[q][2]] + cw[q][3] * hp[coff[q][3]];
        A1[pix * K1B + k] = f2bf(v);
    }
    __syncthreads();

    // --- stage 3: GEMM1 (deform, K=1152) -> ReLU -> fused w_bb tail (K=128) --
    const int coutn = wave * 16 + lo;
    const char* brow = (const char*)wregbb + (size_t)coutn * (K1B * 2);
    v8f acc = {};
    for (int ks = 0; ks < 36; ++ks) {          // K = 0..1151
        const int k0 = ks * 32;
        Frag fa, fb;
        fa.q[0] = *(const u32x4*)(&A1[lo * K1B + k0 + half * 8]);
        fa.q[1] = *(const u32x4*)(&A1[lo * K1B + k0 + 16 + half * 8]);
        const u32x4* bp = (const u32x4*)(brow + (size_t)(k0 + half * 8) * 2);
        fb.q[0] = bp[0];
        fb.q[1] = bp[2];
        acc = __builtin_amdgcn_wmma_f32_16x16x32_bf16(false, fa.v, false, fb.v,
                                                      (short)0, acc, false, false);
    }
#pragma unroll
    for (int r = 0; r < 8; ++r) acc[r] = fmaxf(acc[r], 0.0f);   // ReLU(deform)
    for (int ks = 36; ks < 40; ++ks) {         // K = 1152..1279 : w_bb * DSMg
        const int k0 = ks * 32;
        Frag fa, fb;
        fa.q[0] = *(const u32x4*)(&A1[lo * K1B + k0 + half * 8]);
        fa.q[1] = *(const u32x4*)(&A1[lo * K1B + k0 + 16 + half * 8]);
        const u32x4* bp = (const u32x4*)(brow + (size_t)(k0 + half * 8) * 2);
        fb.q[0] = bp[0];
        fb.q[1] = bp[2];
        acc = __builtin_amdgcn_wmma_f32_16x16x32_bf16(false, fa.v, false, fb.v,
                                                      (short)0, acc, false, false);
    }
    const float bbias = b_bb[coutn];
#pragma unroll
    for (int r = 0; r < 8; ++r) acc[r] += bbias;

    // --- stage 4: s tile -> A2 head (D layout: N=lo, M=r+half*8) -------------
#pragma unroll
    for (int r = 0; r < 8; ++r) {
        const int m = r + half * 8;
        A2[m * K2 + coutn] = f2bf(acc[r]);
    }
    __syncthreads();

    // --- stage 5: GEMM2 (concat 1x1, K=256) + bias + ReLU + store ------------
    const char* brow2 = (const char*)wshare + (size_t)coutn * (K2 * 2);
    v8f acc2 = {};
#pragma unroll
    for (int ks = 0; ks < 8; ++ks) {
        const int k0 = ks * 32;
        Frag fa, fb;
        fa.q[0] = *(const u32x4*)(&A2[lo * K2 + k0 + half * 8]);
        fa.q[1] = *(const u32x4*)(&A2[lo * K2 + k0 + 16 + half * 8]);
        const u32x4* bp = (const u32x4*)(brow2 + (size_t)(k0 + half * 8) * 2);
        fb.q[0] = bp[0];
        fb.q[1] = bp[2];
        acc2 = __builtin_amdgcn_wmma_f32_16x16x32_bf16(false, fa.v, false, fb.v,
                                                       (short)0, acc2, false, false);
    }
    const float bs = b_share[coutn];
    float* orow = out + ((size_t)(b * 128 + coutn) * Hh + y) * Ww + x0 + half * 8;
    f32x4 st0, st1;
    st0.x = fmaxf(acc2[0] + bs, 0.f); st0.y = fmaxf(acc2[1] + bs, 0.f);
    st0.z = fmaxf(acc2[2] + bs, 0.f); st0.w = fmaxf(acc2[3] + bs, 0.f);
    st1.x = fmaxf(acc2[4] + bs, 0.f); st1.y = fmaxf(acc2[5] + bs, 0.f);
    st1.z = fmaxf(acc2[6] + bs, 0.f); st1.w = fmaxf(acc2[7] + bs, 0.f);
    *(f32x4*)(orow)     = st0;        // two global_store_b128 per lane
    *(f32x4*)(orow + 4) = st1;
}

// ---------------------------------------------------------------------------
extern "C" void kernel_launch(void* const* d_in, const int* in_sizes, int n_in,
                              void* d_out, int out_size, void* d_ws, size_t ws_size,
                              hipStream_t stream) {
    (void)in_sizes; (void)n_in; (void)out_size; (void)ws_size;
    const float* HSI     = (const float*)d_in[0];
    const float* DSM     = (const float*)d_in[1];
    const float* SHR     = (const float*)d_in[2];
    const float* w_off   = (const float*)d_in[3];
    const float* b_off   = (const float*)d_in[4];
    const float* w_mask  = (const float*)d_in[5];
    const float* b_mask  = (const float*)d_in[6];
    const float* w_reg   = (const float*)d_in[7];
    const float* w_bb    = (const float*)d_in[8];
    const float* b_bb    = (const float*)d_in[9];
    const float* w_share = (const float*)d_in[10];
    const float* b_share = (const float*)d_in[11];
    const float* lam1    = (const float*)d_in[12];
    const float* lam2    = (const float*)d_in[13];

    char* ws = (char*)d_ws;
    unsigned short* wregbb = (unsigned short*)(ws + WS_WREGBB);
    unsigned short* wshare = (unsigned short*)(ws + WS_WSHARE);
    unsigned short* woffm  = (unsigned short*)(ws + WS_WOFFM);
    float*          offsb  = (float*)(ws + WS_OFFS);
    float*          maskb  = (float*)(ws + WS_MASK);

    // 0) pack weights to bf16 k-major layouts (one-shot, tiny)
    const int npack = 128 * 1280 + 128 * 256 + 27 * 1152;
    pack_weights_kernel<<<(npack + 255) / 256, 256, 0, stream>>>(
        w_off, w_mask, w_reg, w_bb, w_share, wregbb, wshare, woffm);

    // 1) offsets + sigmoid(mask) conv3x3 over DSMg (recomputed on the fly)
    offsets_mask_kernel<<<(Bm * HW) / 256, 256, 0, stream>>>(
        HSI, DSM, b_off, b_mask, lam1, lam2, woffm, offsb, maskb);

    // 2) fused deform-conv + ReLU + w_bb 1x1 + concat-1x1 + bias + ReLU
    deform_fused_kernel<<<Bm * Hh * (Ww / 16), 256, 0, stream>>>(
        HSI, DSM, SHR, b_bb, b_share, lam1, lam2, wregbb, wshare,
        offsb, maskb, (float*)d_out);
}
